// MultiHeadAttention_18494129176737
// MI455X (gfx1250) — matrix operand order repaired and checked by reading, not compiled, not run on hardware
//
#include <hip/hip_runtime.h>
#include <cstddef>

// ---------------------------------------------------------------------------
// MHA for MI455X (gfx1250, wave32, WMMA 16x16x32 f16 / f32-accum).
// Faithful to the reference quirks: scale = 1/sqrt(NHEAD) = 0.25 and softmax
// over the QUERY axis (per-key-column normalization).
// ---------------------------------------------------------------------------

#define S_LEN 2048
#define DM    1024
#define NH    16
#define HD    64
#define MROWS 4096          // B * S

typedef __attribute__((ext_vector_type(16))) _Float16 v16h;
typedef __attribute__((ext_vector_type(8)))  _Float16 v8h;
typedef __attribute__((ext_vector_type(8)))  float    v8f;
typedef __attribute__((ext_vector_type(4)))  int      v4i;

#if __has_builtin(__builtin_amdgcn_global_load_async_to_lds_b128) && \
    __has_builtin(__builtin_amdgcn_s_wait_asynccnt)
#define HAVE_ASYNC_LDS 1
#else
#define HAVE_ASYNC_LDS 0
#endif

#if HAVE_ASYNC_LDS
typedef __attribute__((address_space(1))) v4i gv4i;   // global v4i
typedef __attribute__((address_space(3))) v4i lv4i;   // LDS v4i
// DMA 16 bytes global -> LDS per lane (GLOBAL_LOAD_ASYNC_TO_LDS_B128, ASYNCcnt).
static __device__ __forceinline__ void async_b128(const void* g, void* l) {
    __builtin_amdgcn_global_load_async_to_lds_b128(
        (gv4i*)(unsigned long long)g,
        (lv4i*)(unsigned)(unsigned long long)l,
        0, 0);
}
static __device__ __forceinline__ void async_wait0() {
    __builtin_amdgcn_s_wait_asynccnt(0);
}
#endif

static __device__ __forceinline__ v8f v8f_zero() {
    v8f z;
#pragma unroll
    for (int i = 0; i < 8; ++i) z[i] = 0.0f;
    return z;
}

static __device__ __forceinline__ v8f wmma_f16(v16h a, v16h b, v8f c) {
    // (neg_a, A, neg_b, B, c_mod, C, reuse_a, reuse_b)
    return __builtin_amdgcn_wmma_f32_16x16x32_f16(false, a, false, b,
                                                  (short)0, c, false, false);
}

// A fragment (16x32 f16, MxK) from a row-major LDS tile [rows][stride halves].
// Lane l: M = m0 + l%16, K = k0 + base + {0..7} and k0 + base+16 + {0..7},
// base = 8*(l>=16).  Two aligned 16-byte LDS loads.
static __device__ __forceinline__ v16h ldsA_frag(const _Float16* t, int stride,
                                                 int m0, int k0) {
    const int l   = threadIdx.x & 31;
    const int row = m0 + (l & 15);
    const int kb  = k0 + ((l >> 4) << 3);
    const _Float16* p = t + row * stride + kb;
    v8h lo = *(const v8h*)p;
    v8h hi = *(const v8h*)(p + 16);
    v16h r;
#pragma unroll
    for (int i = 0; i < 8; ++i) { r[i] = lo[i]; r[i + 8] = hi[i]; }
    return r;
}

// B fragment (32x16 f16, KxN) from an [n][k] LDS tile (k contiguous).
// Lane l: N = n0 + l%16, K = k0 + base + {0..7} and k0 + base+16 + {0..7}, base = 8*(l>=16).
static __device__ __forceinline__ v16h ldsB_frag(const _Float16* t, int stride,
                                                 int n0, int k0) {
    const int l  = threadIdx.x & 31;
    const int n  = n0 + (l & 15);
    const int kb = k0 + ((l >> 4) << 3);
    const _Float16* p = t + n * stride + kb;
    v8h lo = *(const v8h*)p;
    v8h hi = *(const v8h*)(p + 16);
    v16h r;
#pragma unroll
    for (int i = 0; i < 8; ++i) { r[i] = lo[i]; r[i + 8] = hi[i]; }
    return r;
}

// ---------------------------------------------------------------------------
// Pack mask[2048][2048] (int 0/1) into bit words: mbits[i][w] bit b =
// mask[i][w*32+b].  16 MB -> 512 KB (L2-resident, read by every tile).
// ---------------------------------------------------------------------------
__global__ __launch_bounds__(256) void mask_pack_kernel(
    const int* __restrict__ mask, unsigned* __restrict__ mbits) {
    const int idx = blockIdx.x * 256 + threadIdx.x;   // i*64 + wj
    const int i = idx >> 6, wj = idx & 63;
    const int4* src = (const int4*)(mask + (size_t)i * S_LEN + wj * 32);
    unsigned m = 0;
#pragma unroll
    for (int q = 0; q < 8; ++q) {
        int4 v = src[q];
        m |= (v.x ? 1u : 0u) << (4 * q + 0);
        m |= (v.y ? 1u : 0u) << (4 * q + 1);
        m |= (v.z ? 1u : 0u) << (4 * q + 2);
        m |= (v.w ? 1u : 0u) << (4 * q + 3);
    }
    mbits[idx] = m;
}

// ---------------------------------------------------------------------------
// GEMM:  C[M][N] = A[M][K] @ W[N][K]^T + bias[N]      (out = X @ W.T + b)
// Tile 128x128x32, 8 waves as 2(M)x4(N), each wave 64x32 = 4x2 WMMA frags.
// AT = float (convert to f16 on the fly) or _Float16 (async DMA to LDS).
// ---------------------------------------------------------------------------
template <typename AT, typename OT>
__global__ __launch_bounds__(256) void gemm_bias_kernel(
    const AT* __restrict__ A, const float* __restrict__ W,
    const float* __restrict__ bias, OT* __restrict__ C,
    int M, int N, int K) {
    __shared__ __align__(16) _Float16 As[128 * 32];
    __shared__ __align__(16) _Float16 Bs[128 * 32];

    const int tid = threadIdx.x;
    const int m0  = blockIdx.y * 128;
    const int n0  = blockIdx.x * 128;
    const int w   = tid >> 5;
    const int wm  = w >> 2;            // 0..1
    const int wn  = w & 3;             // 0..3
    const int ln  = tid & 15;
    const int lh  = (tid >> 4) & 1;

    v8f c[4][2];
#pragma unroll
    for (int i = 0; i < 4; ++i)
#pragma unroll
        for (int j = 0; j < 2; ++j) c[i][j] = v8f_zero();

    const int row = tid >> 1;
    const int cb  = (tid & 1) * 16;

    for (int k0 = 0; k0 < K; k0 += 32) {
        // ---- stage A tile [128 m][32 k] ----
        if constexpr (sizeof(AT) == 4) {
            const float4* src = (const float4*)(A + (size_t)(m0 + row) * K + k0 + cb);
            _Float16* dst = &As[row * 32 + cb];
#pragma unroll
            for (int q = 0; q < 4; ++q) {
                float4 f = src[q];
                dst[4 * q + 0] = (_Float16)f.x; dst[4 * q + 1] = (_Float16)f.y;
                dst[4 * q + 2] = (_Float16)f.z; dst[4 * q + 3] = (_Float16)f.w;
            }
        } else {
#if HAVE_ASYNC_LDS
            async_b128(A + (size_t)(m0 + row) * K + k0 + cb,     &As[row * 32 + cb]);
            async_b128(A + (size_t)(m0 + row) * K + k0 + cb + 8, &As[row * 32 + cb + 8]);
#else
            const v8h* src = (const v8h*)(A + (size_t)(m0 + row) * K + k0 + cb);
            *(v8h*)&As[row * 32 + cb]     = src[0];
            *(v8h*)&As[row * 32 + cb + 8] = src[1];
#endif
        }
        // ---- stage B tile [128 n][32 k] from W rows (coalesced) ----
        {
            const float4* src = (const float4*)(W + (size_t)(n0 + row) * K + k0 + cb);
            _Float16* dst = &Bs[row * 32 + cb];
#pragma unroll
            for (int q = 0; q < 4; ++q) {
                float4 f = src[q];
                dst[4 * q + 0] = (_Float16)f.x; dst[4 * q + 1] = (_Float16)f.y;
                dst[4 * q + 2] = (_Float16)f.z; dst[4 * q + 3] = (_Float16)f.w;
            }
        }
#if HAVE_ASYNC_LDS
        if constexpr (sizeof(AT) == 2) async_wait0();
#endif
        __syncthreads();

        v16h a[4], bf[2];
#pragma unroll
        for (int mt = 0; mt < 4; ++mt) a[mt] = ldsA_frag(As, 32, wm * 64 + mt * 16, 0);
#pragma unroll
        for (int nt = 0; nt < 2; ++nt) bf[nt] = ldsB_frag(Bs, 32, wn * 32 + nt * 16, 0);
#pragma unroll
        for (int mt = 0; mt < 4; ++mt)
#pragma unroll
            for (int nt = 0; nt < 2; ++nt) c[mt][nt] = wmma_f16(a[mt], bf[nt], c[mt][nt]);
        __syncthreads();
    }

#pragma unroll
    for (int nt = 0; nt < 2; ++nt) {
        const int n  = n0 + wn * 32 + nt * 16 + ln;
        const float bv = bias[n];
#pragma unroll
        for (int mt = 0; mt < 4; ++mt)
#pragma unroll
            for (int r = 0; r < 8; ++r) {
                const int m = m0 + wm * 64 + mt * 16 + r + 8 * lh;
                C[(size_t)m * N + n] = (OT)(c[mt][nt][r] + bv);
            }
    }
}

// ---------------------------------------------------------------------------
// Pass 1: per-key-column sums of exp(0.25*score) with masking.
// One WG = one (bh, iBlock=128, jBlock=128) tile; K = 64 (head dim).
// Writes partial[bh][iBlock][j] (deterministic fixed-order reduction later).
// ---------------------------------------------------------------------------
__global__ __launch_bounds__(256) void attn_colsum_kernel(
    const _Float16* __restrict__ q16, const _Float16* __restrict__ k16,
    const unsigned* __restrict__ mbits, float* __restrict__ partial) {
    __shared__ __align__(16) _Float16 Qs[128 * 64];
    __shared__ __align__(16) _Float16 Ks[128 * 64];
    __shared__ unsigned Ms[128 * 4];      // mask bits for this 128x128 tile
    __shared__ float colred[2 * 128 * 2];

    const int tid = threadIdx.x;
    const int bh  = blockIdx.z;
    const int b   = bh >> 4, h = bh & 15;
    const int i0  = blockIdx.y * 128, j0 = blockIdx.x * 128;
    const _Float16* qh = q16 + (size_t)b * S_LEN * DM + h * HD;
    const _Float16* kh = k16 + (size_t)b * S_LEN * DM + h * HD;

    {
        const int row = tid >> 1;
        const int cb  = (tid & 1) * 32;
#if HAVE_ASYNC_LDS
#pragma unroll
        for (int q = 0; q < 4; ++q) {
            async_b128(qh + (size_t)(i0 + row) * DM + cb + 8 * q, &Qs[row * 64 + cb + 8 * q]);
            async_b128(kh + (size_t)(j0 + row) * DM + cb + 8 * q, &Ks[row * 64 + cb + 8 * q]);
        }
#else
        const v8h* sq = (const v8h*)(qh + (size_t)(i0 + row) * DM + cb);
        const v8h* sk = (const v8h*)(kh + (size_t)(j0 + row) * DM + cb);
#pragma unroll
        for (int q = 0; q < 4; ++q) {
            *(v8h*)&Qs[row * 64 + cb + 8 * q] = sq[q];
            *(v8h*)&Ks[row * 64 + cb + 8 * q] = sk[q];
        }
#endif
        // mask bit tile: 128 rows x 4 words
#pragma unroll
        for (int t = tid; t < 512; t += 256)
            Ms[t] = mbits[(size_t)(i0 + (t >> 2)) * 64 + (j0 >> 5) + (t & 3)];
    }
#if HAVE_ASYNC_LDS
    async_wait0();
#endif
    __syncthreads();

    const int w = tid >> 5, wm = w >> 2, wn = w & 3;
    const int ln = tid & 15, lh = (tid >> 4) & 1;

    v8f c[4][2];
#pragma unroll
    for (int i = 0; i < 4; ++i)
#pragma unroll
        for (int j = 0; j < 2; ++j) c[i][j] = v8f_zero();

#pragma unroll
    for (int ks = 0; ks < 2; ++ks) {
        const int k0 = ks * 32;
        v16h a[4];
#pragma unroll
        for (int mt = 0; mt < 4; ++mt) a[mt] = ldsA_frag(Qs, 64, wm * 64 + mt * 16, k0);
#pragma unroll
        for (int nt = 0; nt < 2; ++nt) {
            v16h bf = ldsB_frag(Ks, 64, wn * 32 + nt * 16, k0);
#pragma unroll
            for (int mt = 0; mt < 4; ++mt) c[mt][nt] = wmma_f16(a[mt], bf, c[mt][nt]);
        }
    }

    // masked exp + per-lane partial column sums (exp at -10000 == 0 exactly)
#pragma unroll
    for (int nt = 0; nt < 2; ++nt) {
        const int jl = wn * 32 + nt * 16 + ln;     // local j in tile
        float p = 0.0f;
#pragma unroll
        for (int mt = 0; mt < 4; ++mt)
#pragma unroll
            for (int r = 0; r < 8; ++r) {
                const int il = wm * 64 + mt * 16 + r + 8 * lh;   // local i
                const float s = c[mt][nt][r] * 0.25f;
                const unsigned mb = (Ms[il * 4 + (jl >> 5)] >> (jl & 31)) & 1u;
                p += mb ? __expf(s) : 0.0f;
            }
        colred[(wm * 128 + jl) * 2 + lh] = p;
    }
    __syncthreads();
    if (tid < 128) {
        const float s = colred[(0 * 128 + tid) * 2 + 0] + colred[(0 * 128 + tid) * 2 + 1] +
                        colred[(1 * 128 + tid) * 2 + 0] + colred[(1 * 128 + tid) * 2 + 1];
        partial[((size_t)bh * 16 + blockIdx.y) * S_LEN + j0 + tid] = s;
    }
}

// Fixed-order reduction of the 16 i-block partials -> 1/Z per (bh, j).
__global__ __launch_bounds__(256) void colsum_reduce_kernel(
    const float* __restrict__ partial, float* __restrict__ invZ) {
    const int idx = blockIdx.x * 256 + threadIdx.x;   // bh*2048 + j
    const int bh = idx >> 11, j = idx & 2047;
    float s = 0.0f;
#pragma unroll
    for (int ib = 0; ib < 16; ++ib) s += partial[((size_t)bh * 16 + ib) * S_LEN + j];
    invZ[idx] = 1.0f / s;
}

// ---------------------------------------------------------------------------
// Pass 2: ctx[b,i,h,:] = sum_j (exp(0.25*s[i,j]) * invZ[j] * mask) * v[j,h,:]
// One WG = (bh, iBlock=128).  Streams j in 32-wide tiles: recompute score
// tile with WMMA, transform to f16 probabilities in LDS, then P @ V WMMA.
// ---------------------------------------------------------------------------
__global__ __launch_bounds__(256) void attn_ctx_kernel(
    const _Float16* __restrict__ q16, const _Float16* __restrict__ k16,
    const _Float16* __restrict__ v16, const unsigned* __restrict__ mbits,
    const float* __restrict__ invZ, _Float16* __restrict__ ctx16) {
    __shared__ __align__(16) _Float16 Qs[128 * 64];   // [i][d]
    __shared__ __align__(16) _Float16 Ks[32 * 64];    // [j][d]  (B tile, n=j)
    __shared__ __align__(16) _Float16 Vs[64 * 32];    // [dh][j] (B tile, n=dh)
    __shared__ __align__(16) _Float16 Ps[128 * 32];   // [i][j]  (A tile)
    __shared__ unsigned Ms[128];                      // mask bits, 1 word/row

    const int tid = threadIdx.x;
    const int bh  = blockIdx.y;
    const int b   = bh >> 4, h = bh & 15;
    const int i0  = blockIdx.x * 128;
    const _Float16* qh = q16 + (size_t)b * S_LEN * DM + h * HD;
    const _Float16* kh = k16 + (size_t)b * S_LEN * DM + h * HD;
    const _Float16* vh = v16 + (size_t)b * S_LEN * DM + h * HD;
    const float* izh   = invZ + (size_t)bh * S_LEN;

    {   // Q tile, loaded once
        const int row = tid >> 1;
        const int cb  = (tid & 1) * 32;
#if HAVE_ASYNC_LDS
#pragma unroll
        for (int q = 0; q < 4; ++q)
            async_b128(qh + (size_t)(i0 + row) * DM + cb + 8 * q, &Qs[row * 64 + cb + 8 * q]);
#else
        const v8h* sq = (const v8h*)(qh + (size_t)(i0 + row) * DM + cb);
#pragma unroll
        for (int q = 0; q < 4; ++q) *(v8h*)&Qs[row * 64 + cb + 8 * q] = sq[q];
#endif
    }

    const int w = tid >> 5;
    const int ln = tid & 15, lh = (tid >> 4) & 1;
    const int mq = w >> 1, nq = w & 1;   // PV mapping: 4(M) x 2(N)

    v8f acc[2][2];
#pragma unroll
    for (int i = 0; i < 2; ++i)
#pragma unroll
        for (int j = 0; j < 2; ++j) acc[i][j] = v8f_zero();

    for (int jt = 0; jt < S_LEN / 32; ++jt) {
        const int j0 = jt * 32;
        {   // K tile [32 j][64 d]  (coalesced)
            const int jr  = tid >> 3;
            const int seg = (tid & 7) * 8;
#if HAVE_ASYNC_LDS
            async_b128(kh + (size_t)(j0 + jr) * DM + seg, &Ks[jr * 64 + seg]);
#else
            *(v8h*)&Ks[jr * 64 + seg] =
                *(const v8h*)(kh + (size_t)(j0 + jr) * DM + seg);
#endif
        }
        {   // V tile transposed -> [dh][j]
            const int dh = tid & 63;
            const int jb = (tid >> 6) * 8;
#pragma unroll
            for (int jj = 0; jj < 8; ++jj)
                Vs[dh * 32 + jb + jj] = vh[(size_t)(j0 + jb + jj) * DM + dh];
        }
        if (tid < 128) Ms[tid] = mbits[(size_t)(i0 + tid) * 64 + (j0 >> 5)];
#if HAVE_ASYNC_LDS
        async_wait0();
#endif
        __syncthreads();

        // score tile: M=128 (wave w owns m-tile w), N=32, K=64
        v8f sc[2];
        sc[0] = v8f_zero(); sc[1] = v8f_zero();
#pragma unroll
        for (int ks = 0; ks < 2; ++ks) {
            v16h a = ldsA_frag(Qs, 64, w * 16, ks * 32);
#pragma unroll
            for (int nt = 0; nt < 2; ++nt) {
                v16h bf = ldsB_frag(Ks, 64, nt * 16, ks * 32);
                sc[nt] = wmma_f16(a, bf, sc[nt]);
            }
        }
        // probabilities -> f16 A tile in LDS
#pragma unroll
        for (int nt = 0; nt < 2; ++nt) {
            const int jl = nt * 16 + ln;          // local j (0..31)
            const float iz = izh[j0 + jl];
#pragma unroll
            for (int r = 0; r < 8; ++r) {
                const int il = w * 16 + r + 8 * lh;   // local i
                const float s = sc[nt][r] * 0.25f;
                const unsigned mb = (Ms[il] >> (jl & 31)) & 1u;
                const float e = mb ? __expf(s) * iz : 0.0f;
                Ps[il * 32 + jl] = (_Float16)e;
            }
        }
        __syncthreads();

        // P @ V : M=128, N=64, K=32; wave = 32x32 region (2x2 frags)
        v16h pa[2], vb[2];
#pragma unroll
        for (int mt = 0; mt < 2; ++mt) pa[mt] = ldsA_frag(Ps, 32, mq * 32 + mt * 16, 0);
#pragma unroll
        for (int nt = 0; nt < 2; ++nt) vb[nt] = ldsB_frag(Vs, 32, nq * 32 + nt * 16, 0);
#pragma unroll
        for (int mt = 0; mt < 2; ++mt)
#pragma unroll
            for (int nt = 0; nt < 2; ++nt) acc[mt][nt] = wmma_f16(pa[mt], vb[nt], acc[mt][nt]);
        __syncthreads();
    }

#pragma unroll
    for (int mt = 0; mt < 2; ++mt)
#pragma unroll
        for (int nt = 0; nt < 2; ++nt)
#pragma unroll
            for (int r = 0; r < 8; ++r) {
                const int ii = i0 + mq * 32 + mt * 16 + r + 8 * lh;
                const int dh = nq * 32 + nt * 16 + ln;
                ctx16[(size_t)(b * S_LEN + ii) * DM + h * HD + dh] = (_Float16)acc[mt][nt][r];
            }
}

// ---------------------------------------------------------------------------
extern "C" void kernel_launch(void* const* d_in, const int* in_sizes, int n_in,
                              void* d_out, int out_size, void* d_ws, size_t ws_size,
                              hipStream_t stream) {
    (void)in_sizes; (void)n_in; (void)out_size; (void)ws_size;
    const float* query = (const float*)d_in[0];
    const float* key_  = (const float*)d_in[1];
    const float* value = (const float*)d_in[2];
    const int*   mask  = (const int*)d_in[3];
    const float* Wq = (const float*)d_in[4];  const float* bq = (const float*)d_in[5];
    const float* Wk = (const float*)d_in[6];  const float* bk = (const float*)d_in[7];
    const float* Wv = (const float*)d_in[8];  const float* bv = (const float*)d_in[9];
    const float* Wp = (const float*)d_in[10]; const float* bp = (const float*)d_in[11];
    float* out = (float*)d_out;

    // workspace layout (needs ~38.6 MB)
    char* wsb = (char*)d_ws;
    _Float16* q16   = (_Float16*)(wsb + (size_t) 0 * 1024 * 1024);
    _Float16* k16   = (_Float16*)(wsb + (size_t) 8 * 1024 * 1024);
    _Float16* v16   = (_Float16*)(wsb + (size_t)16 * 1024 * 1024);
    _Float16* ctx16 = (_Float16*)(wsb + (size_t)24 * 1024 * 1024);
    float*    part  = (float*)   (wsb + (size_t)32 * 1024 * 1024);
    float*    invZ  = (float*)   (wsb + (size_t)36 * 1024 * 1024);
    unsigned* mbits = (unsigned*)(wsb + (size_t)36 * 1024 * 1024 + 512 * 1024);

    const dim3 blk(256);
    const dim3 gproj(DM / 128, MROWS / 128);                 // 8 x 32

    mask_pack_kernel<<<dim3((S_LEN * 64) / 256), blk, 0, stream>>>(mask, mbits);

    gemm_bias_kernel<float, _Float16><<<gproj, blk, 0, stream>>>(query, Wq, bq, q16, MROWS, DM, DM);
    gemm_bias_kernel<float, _Float16><<<gproj, blk, 0, stream>>>(key_,  Wk, bk, k16, MROWS, DM, DM);
    gemm_bias_kernel<float, _Float16><<<gproj, blk, 0, stream>>>(value, Wv, bv, v16, MROWS, DM, DM);

    attn_colsum_kernel<<<dim3(S_LEN / 128, S_LEN / 128, 2 * NH), blk, 0, stream>>>(q16, k16, mbits, part);
    colsum_reduce_kernel<<<dim3((2 * NH * S_LEN) / 256), blk, 0, stream>>>(part, invZ);
    attn_ctx_kernel<<<dim3(S_LEN / 128, 2 * NH), blk, 0, stream>>>(q16, k16, v16, mbits, invZ, ctx16);

    gemm_bias_kernel<_Float16, float><<<gproj, blk, 0, stream>>>(ctx16, Wp, bp, out, MROWS, DM, DM);
}
